// CskCauseDag0_24979529793778
// MI455X (gfx1250) — matrix operand order repaired and checked by reading, not compiled, not run on hardware
//
#include <hip/hip_runtime.h>
#include <hip/hip_bf16.h>
#include <stdint.h>

// ===========================================================================
// MI455X (gfx1250) implementation of the graph-GRU ("GRNK") reference.
//
// Restructuring vs. reference (math-equivalent up to bf16 GEMM rounding):
//  * M = (sum_n w*s*H_n) @ wr0.T + (sum_n w*o*H_n) @ wr1.T  (reduce-then-GEMM,
//    folded into ONE GEMM by concatenating along K: [u0|u1] @ [wr0|wr1]^T)
//  * state-independent GRU halves precomputed as big batched WMMA GEMMs
//  * attention logit = per-step scalar (f_i . att_q) + cached (h_n . att_h)
// GEMMs: v_wmma_f32_16x16x32_bf16. K loop unrolled x2 with ping-pong fragment
// buffers (no loop-carried register copies) AND strength-reduced addressing:
// one base pointer per load stream advanced by a constant 128 B/iter, all
// loads via immediate offsets (avoids 64-bit address recomputation -> no
// scratch spills). Last iteration peeled so the steady-state body is
// branch-free. K must be a multiple of 64 (call sites: 512, 1024, 1536).
// Requires ws_size >= ~64 MB.
// ===========================================================================

#define DEV __device__ __forceinline__

typedef __attribute__((ext_vector_type(16))) __bf16 v16bf;
typedef __attribute__((ext_vector_type(8)))  float  v8f;

constexpr int Bb = 16, Nn = 64, Dd = 512, Ll = 2;
constexpr int D3   = 3 * Dd;          // 1536
constexpr int CATD = (Ll + 1) * Dd;   // 1536

union FragAB { v16bf v; uint4 q[2]; };
union FragC  { v8f  v; float f[8]; };

DEV __bf16 to_bf16(float x) { return (__bf16)x; }
DEV float  sigm(float x)    { return 1.0f / (1.0f + __expf(-x)); }

// C/D f32 16x16: VGPR r -> row m0+r+8*half, col n0+(lane&15).
DEV void store_fragC(float* __restrict__ Y, __bf16* __restrict__ Ybf, int ldy,
                     int m0, int n0, int lane, v8f acc, const float* __restrict__ bias) {
  int n = lane & 15, half = lane >> 4;
  float bv = bias ? bias[n0 + n] : 0.0f;
  FragC c; c.v = acc;
  #pragma unroll
  for (int r = 0; r < 8; ++r) {
    size_t idx = (size_t)(m0 + r + 8 * half) * ldy + (n0 + n);
    float v = c.f[r] + bv;
    Y[idx] = v;
    if (Ybf) Ybf[idx] = to_bf16(v);
  }
}

// ---- Generic GEMM: Y[M,N] = Xbf[M,K] @ Wbf[N,K]^T (+bias[N]) --------------
// One wave computes a 16x(16*NT) output tile.
// A-frag layout (ISA 7.12.2, 16-bit A 16x32): lane m=lane&15, half=lane>>4;
//   per-lane runs at bytes +0..15 (K=k+8h..) and +32..47 (K=k+16+8h..)
//   relative to pa = X + (m0+m)*K + 8*half.
// B-frag (B=W^T): lane holds W row n0+n, same K runs as A at pb = W row + 8*half (+0..7, +16..23).
template <int NT>
__global__ void wmma_gemm_xwT(const __bf16* __restrict__ X, const __bf16* __restrict__ W,
                              const float* __restrict__ bias, float* __restrict__ Y,
                              __bf16* __restrict__ Ybf, int M, int N, int K) {
  int lane = threadIdx.x & 31;
  int m    = lane & 15;
  int half = lane >> 4;
  int gw   = (int)((blockIdx.x * blockDim.x + threadIdx.x) >> 5);
  int tn   = N / (16 * NT);
  int mt   = gw / tn;
  if (mt * 16 >= M) return;              // whole-wave uniform: EXEC stays all-1s
  int n0 = (gw - mt * tn) * (16 * NT);
  int m0 = mt << 4;

  // per-stream base pointers; advance by 64 elements (128 B) per iteration
  const __bf16* pa = X + (size_t)(m0 + m) * K + half * 8;
  const __bf16* pb[NT];
  #pragma unroll
  for (int t = 0; t < NT; ++t) pb[t] = W + (size_t)(n0 + 16 * t + m) * K + half * 8;

  v8f acc[NT];
  #pragma unroll
  for (int t = 0; t < NT; ++t) acc[t] = (v8f){};

  FragAB A0, A1, B0[NT], B1[NT];
  // prologue: buffer 0 holds k = 0
  A0.q[0] = *reinterpret_cast<const uint4*>(pa);
  A0.q[1] = *reinterpret_cast<const uint4*>(pa + 16);
  #pragma unroll
  for (int t = 0; t < NT; ++t) {
    B0[t].q[0] = *reinterpret_cast<const uint4*>(pb[t]);
    B0[t].q[1] = *reinterpret_cast<const uint4*>(pb[t] + 16);
  }

  int iters = K >> 6;                    // K/64, >= 8 at every call site
  for (int it = 0; it < iters - 1; ++it) {
    // load buffer 1 (k+32) via immediate offsets, then consume buffer 0 (k)
    A1.q[0] = *reinterpret_cast<const uint4*>(pa + 32);
    A1.q[1] = *reinterpret_cast<const uint4*>(pa + 48);
    #pragma unroll
    for (int t = 0; t < NT; ++t) {
      B1[t].q[0] = *reinterpret_cast<const uint4*>(pb[t] + 32);
      B1[t].q[1] = *reinterpret_cast<const uint4*>(pb[t] + 48);
    }
    #pragma unroll
    for (int t = 0; t < NT; ++t)
      acc[t] = __builtin_amdgcn_wmma_f32_16x16x32_bf16(false, A0.v, false, B0[t].v,
                                                       (short)0, acc[t], false, false);
    // advance streams, load buffer 0 (k+64), consume buffer 1 (k+32)
    pa += 64;
    #pragma unroll
    for (int t = 0; t < NT; ++t) pb[t] += 64;
    A0.q[0] = *reinterpret_cast<const uint4*>(pa);
    A0.q[1] = *reinterpret_cast<const uint4*>(pa + 16);
    #pragma unroll
    for (int t = 0; t < NT; ++t) {
      B0[t].q[0] = *reinterpret_cast<const uint4*>(pb[t]);
      B0[t].q[1] = *reinterpret_cast<const uint4*>(pb[t] + 16);
    }
    #pragma unroll
    for (int t = 0; t < NT; ++t)
      acc[t] = __builtin_amdgcn_wmma_f32_16x16x32_bf16(false, A1.v, false, B1[t].v,
                                                       (short)0, acc[t], false, false);
  }
  // peeled last iteration (no buffer-0 refill)
  A1.q[0] = *reinterpret_cast<const uint4*>(pa + 32);
  A1.q[1] = *reinterpret_cast<const uint4*>(pa + 48);
  #pragma unroll
  for (int t = 0; t < NT; ++t) {
    B1[t].q[0] = *reinterpret_cast<const uint4*>(pb[t] + 32);
    B1[t].q[1] = *reinterpret_cast<const uint4*>(pb[t] + 48);
  }
  #pragma unroll
  for (int t = 0; t < NT; ++t)
    acc[t] = __builtin_amdgcn_wmma_f32_16x16x32_bf16(false, A0.v, false, B0[t].v,
                                                     (short)0, acc[t], false, false);
  #pragma unroll
  for (int t = 0; t < NT; ++t)
    acc[t] = __builtin_amdgcn_wmma_f32_16x16x32_bf16(false, A1.v, false, B1[t].v,
                                                     (short)0, acc[t], false, false);

  #pragma unroll
  for (int t = 0; t < NT; ++t)
    store_fragC(Y, Ybf, N, m0, n0 + 16 * t, lane, acc[t], bias);
}

// ---- Per-step attention: masked softmax + weighted state reductions -------
__global__ void step_attn_kernel(const float* __restrict__ F, const float* __restrict__ H,
                                 const float* __restrict__ hdot,
                                 const int* __restrict__ adj, const int* __restrict__ smask,
                                 const int* __restrict__ omask,
                                 const float* __restrict__ att_w, const float* __restrict__ att_b,
                                 __bf16* __restrict__ ucat, int i) {
  int b = blockIdx.x, tid = threadIdx.x;
  __shared__ float red[256];
  __shared__ float wsm[Nn];
  __shared__ int   sS[Nn], sO[Nn];
  const float* f = F + ((size_t)b * Nn + i) * Dd;
  // qdot = f_i . att_w[0:D]
  float p = 0.f;
  for (int d = tid; d < Dd; d += 256) p += f[d] * att_w[d];
  red[tid] = p; __syncthreads();
  for (int s = 128; s > 0; s >>= 1) { if (tid < s) red[tid] += red[tid + s]; __syncthreads(); }
  if (tid < Nn) {
    int n = tid;
    size_t ro = ((size_t)b * Nn + i) * Nn + n;
    sS[n] = smask[ro];
    sO[n] = omask[ro];
    int valid = (n < i) && (adj[ro] != 0);
    wsm[n] = valid ? (red[0] + att_b[0] + hdot[b * Nn + n]) : -3.0e38f;
  }
  __syncthreads();
  if (tid == 0) {   // adj always has the k=-1 diagonal, so >=1 valid entry
    float mx = -3.0e38f;
    for (int n = 0; n < Nn; ++n) mx = fmaxf(mx, wsm[n]);
    float se = 0.f;
    for (int n = 0; n < Nn; ++n) {
      float e = (wsm[n] > -1.0e37f) ? __expf(wsm[n] - mx) : 0.0f;
      wsm[n] = e; se += e;
    }
    float inv = 1.0f / se;
    for (int n = 0; n < Nn; ++n) wsm[n] *= inv;
  }
  __syncthreads();
  // ucat[b] = [ sum_n w*s*H_n | sum_n w*o*H_n ]  (only n<i contribute)
  for (int d = tid; d < Dd; d += 256) {
    float a0 = 0.f, a1 = 0.f;
    for (int n = 0; n < i; ++n) {
      float w = wsm[n];
      if (w != 0.f) {
        float hv = H[((size_t)b * Nn + n) * Dd + d];
        a0 += w * (sS[n] ? hv : 0.f);
        a1 += w * (sO[n] ? hv : 0.f);
      }
    }
    ucat[(size_t)b * 2 * Dd + d]      = to_bf16(a0);
    ucat[(size_t)b * 2 * Dd + Dd + d] = to_bf16(a1);
  }
}

// ---- Per-step elementwise GRU-triple combine ------------------------------
// g = [Mvec@w_hh_c^T+b_hh_c | Mvec@w_ih_p^T+b_ih_p]  (16 x 3072)
__global__ void step_combine_kernel(const float* __restrict__ g,
                                    const float* __restrict__ gi_c, const float* __restrict__ gh_p,
                                    const float* __restrict__ gi_s, const float* __restrict__ gh_s,
                                    const float* __restrict__ Mv, const float* __restrict__ F,
                                    const float* __restrict__ att_w,
                                    float* __restrict__ H, __bf16* __restrict__ Hbf,
                                    __bf16* __restrict__ catbf, float* __restrict__ hdot,
                                    int i, int catOff) {
  int b = blockIdx.x, tid = threadIdx.x;
  __shared__ float red[256];
  size_t rowi = (size_t)b * Nn + i;
  const float* gic = gi_c + rowi * D3;
  const float* ghp = gh_p + rowi * D3;
  const float* gis = gi_s + rowi * D3;
  const float* ghs = gh_s + rowi * D3;
  const float* ghc = g + (size_t)b * 2 * D3;        // Mvec @ w_hh_c^T + b_hh_c
  const float* gip = ghc + D3;                      // Mvec @ w_ih_p^T + b_ih_p
  float dotacc = 0.f;
  for (int d = tid; d < Dd; d += 256) {
    float m  = Mv[b * Dd + d];
    float fv = F[rowi * Dd + d];
    // gru_c(x=f_i, h=M)
    float r = sigm(gic[d] + ghc[d]);
    float z = sigm(gic[Dd + d] + ghc[Dd + d]);
    float n = tanhf(gic[2 * Dd + d] + r * ghc[2 * Dd + d]);
    float hc = (1.f - z) * n + z * m;
    // gru_p(x=M, h=f_i)
    r = sigm(gip[d] + ghp[d]);
    z = sigm(gip[Dd + d] + ghp[Dd + d]);
    n = tanhf(gip[2 * Dd + d] + r * ghp[2 * Dd + d]);
    float hp = (1.f - z) * n + z * fv;
    // gru_s(x=k_i, h=f_i)
    r = sigm(gis[d] + ghs[d]);
    z = sigm(gis[Dd + d] + ghs[Dd + d]);
    n = tanhf(gis[2 * Dd + d] + r * ghs[2 * Dd + d]);
    float hs = (1.f - z) * n + z * fv;
    float h = hc + hp + hs;
    H[rowi * Dd + d]   = h;
    Hbf[rowi * Dd + d] = to_bf16(h);
    catbf[rowi * CATD + catOff + d] = to_bf16(h);
    dotacc += h * att_w[Dd + d];                    // cache h . att_h
  }
  red[tid] = dotacc; __syncthreads();
  for (int s = 128; s > 0; s >>= 1) { if (tid < s) red[tid] += red[tid + s]; __syncthreads(); }
  if (tid == 0) hdot[b * Nn + i] = red[0];
}

// ---- Step 0: h0 = gru_c(f0,0) + gru_p(0,f0) + gru_s(k0,f0) ----------------
__global__ void init_step_kernel(const float* __restrict__ gi_c, const float* __restrict__ gh_p,
                                 const float* __restrict__ gi_s, const float* __restrict__ gh_s,
                                 const float* __restrict__ b_hh_c, const float* __restrict__ b_ih_p,
                                 const float* __restrict__ F, const float* __restrict__ att_w,
                                 float* __restrict__ H, __bf16* __restrict__ Hbf,
                                 __bf16* __restrict__ catbf, float* __restrict__ hdot, int catOff) {
  int b = blockIdx.x, tid = threadIdx.x;
  __shared__ float red[256];
  size_t rowi = (size_t)b * Nn;                     // i = 0
  const float* gic = gi_c + rowi * D3;
  const float* ghp = gh_p + rowi * D3;
  const float* gis = gi_s + rowi * D3;
  const float* ghs = gh_s + rowi * D3;
  float dotacc = 0.f;
  for (int d = tid; d < Dd; d += 256) {
    float fv = F[rowi * Dd + d];
    // gru_c(f0, 0): gh = b_hh_c only, h = 0
    float r = sigm(gic[d] + b_hh_c[d]);
    float z = sigm(gic[Dd + d] + b_hh_c[Dd + d]);
    float n = tanhf(gic[2 * Dd + d] + r * b_hh_c[2 * Dd + d]);
    float hc = (1.f - z) * n;
    // gru_p(0, f0): gi = b_ih_p only
    r = sigm(b_ih_p[d] + ghp[d]);
    z = sigm(b_ih_p[Dd + d] + ghp[Dd + d]);
    n = tanhf(b_ih_p[2 * Dd + d] + r * ghp[2 * Dd + d]);
    float hp = (1.f - z) * n + z * fv;
    // gru_s(k0, f0)
    r = sigm(gis[d] + ghs[d]);
    z = sigm(gis[Dd + d] + ghs[Dd + d]);
    n = tanhf(gis[2 * Dd + d] + r * ghs[2 * Dd + d]);
    float hs = (1.f - z) * n + z * fv;
    float h = hc + hp + hs;
    H[rowi * Dd + d]   = h;
    Hbf[rowi * Dd + d] = to_bf16(h);
    catbf[rowi * CATD + catOff + d] = to_bf16(h);
    dotacc += h * att_w[Dd + d];
  }
  red[tid] = dotacc; __syncthreads();
  for (int s = 128; s > 0; s >>= 1) { if (tid < s) red[tid] += red[tid + s]; __syncthreads(); }
  if (tid == 0) hdot[b * Nn] = red[0];
}

// ---- Small utility kernels ------------------------------------------------
__global__ void cvt_f32_bf16(const float* __restrict__ s, __bf16* __restrict__ d, size_t n) {
  size_t i = (size_t)blockIdx.x * blockDim.x + threadIdx.x;
  size_t st = (size_t)gridDim.x * blockDim.x;
  for (; i < n; i += st) d[i] = to_bf16(s[i]);
}
// strided variant: d[r*ldd + c] = bf16(s[r*cols + c])  (for K-concat layouts)
__global__ void cvt_f32_bf16_strided(const float* __restrict__ s, __bf16* __restrict__ d,
                                     int rows, int cols, int ldd) {
  size_t total = (size_t)rows * cols;
  size_t i = (size_t)blockIdx.x * blockDim.x + threadIdx.x;
  size_t st = (size_t)gridDim.x * blockDim.x;
  for (; i < total; i += st) {
    size_t r = i / cols; int c = (int)(i % cols);
    d[r * ldd + c] = to_bf16(s[i]);
  }
}
__global__ void copy_f32(const float* __restrict__ s, float* __restrict__ d, size_t n) {
  size_t i = (size_t)blockIdx.x * blockDim.x + threadIdx.x;
  size_t st = (size_t)gridDim.x * blockDim.x;
  for (; i < n; i += st) d[i] = s[i];
}
__global__ void pack_features_kernel(const float* __restrict__ F, __bf16* __restrict__ fbf,
                                     __bf16* __restrict__ catbf) {
  size_t total = (size_t)Bb * Nn * Dd;
  size_t i = (size_t)blockIdx.x * blockDim.x + threadIdx.x;
  size_t st = (size_t)gridDim.x * blockDim.x;
  for (; i < total; i += st) {
    size_t row = i / Dd; int d = (int)(i % Dd);
    __bf16 v = to_bf16(F[i]);
    fbf[i] = v;
    catbf[row * CATD + d] = v;
  }
}
__global__ void gather_know_kernel(const float* __restrict__ know, const int* __restrict__ kadj,
                                   __bf16* __restrict__ kbf) {
  size_t total = (size_t)Bb * Nn * Dd;
  size_t i = (size_t)blockIdx.x * blockDim.x + threadIdx.x;
  size_t st = (size_t)gridDim.x * blockDim.x;
  for (; i < total; i += st) {
    size_t row = i / Dd; int d = (int)(i % Dd);
    int b = (int)(row / Nn), n = (int)(row % Nn);
    int kidx = kadj[((size_t)b * Nn + n) * Nn + n];   // diagonal of know_adj
    kbf[i] = to_bf16(know[(size_t)kidx * Dd + d]);
  }
}

// ---- bf16 weight-block layout (elements, per layer) -----------------------
constexpr size_t MW   = (size_t)D3 * Dd;            // 1536*512
constexpr size_t SW   = (size_t)Dd * Dd;            // 512*512
constexpr size_t W_WHH_C  = 0;                      // gates GEMM: [w_hh_c ; w_ih_p]
constexpr size_t W_WIH_P  = MW;
constexpr size_t W_WIH_C  = 2 * MW;
constexpr size_t W_WHH_P  = 3 * MW;
constexpr size_t W_WIH_S  = 4 * MW;
constexpr size_t W_WHH_S  = 5 * MW;
constexpr size_t W_WRCAT  = 6 * MW;                 // [wr0 | wr1] along K: 512 x 1024
constexpr size_t W_LAYER  = 6 * MW + 2 * SW;        // 5,242,880 elems

// d_in layout: JAX pytree flatten (dict keys sorted alphabetically)
#define L_ATT_B 0
#define L_ATT_W 1
#define L_C_BHH 2
#define L_C_BIH 3
#define L_C_WHH 4
#define L_C_WIH 5
#define L_P_BHH 6
#define L_P_BIH 7
#define L_P_WHH 8
#define L_P_WIH 9
#define L_S_BHH 10
#define L_S_BIH 11
#define L_S_WHH 12
#define L_S_WIH 13
#define L_WR0   14
#define L_WR1   15

template <int NT>
static inline void launch_gemm(const __bf16* X, const __bf16* W, const float* bias,
                               float* Y, __bf16* Ybf, int M, int N, int K, hipStream_t s) {
  int waves = (M / 16) * (N / (16 * NT));
  int blocks = (waves + 7) / 8;
  wmma_gemm_xwT<NT><<<dim3(blocks), dim3(256), 0, s>>>(X, W, bias, Y, Ybf, M, N, K);
}

extern "C" void kernel_launch(void* const* d_in, const int* in_sizes, int n_in,
                              void* d_out, int out_size, void* d_ws, size_t ws_size,
                              hipStream_t stream) {
  (void)in_sizes; (void)n_in; (void)out_size; (void)ws_size;
  const float* features  = (const float*)d_in[0];
  const float* knowledge = (const float*)d_in[1];
  const int*   adj       = (const int*)d_in[2];
  const int*   smask     = (const int*)d_in[3];
  const int*   omask     = (const int*)d_in[4];
  const int*   knowadj   = (const int*)d_in[5];
  const float* emb_b     = (const float*)d_in[6];
  const float* emb_w     = (const float*)d_in[7];
  auto lin = [&](int l, int j) { return (const float*)d_in[8 + 16 * l + j]; };

  // ---- workspace carving (all 16B aligned) ----
  __bf16* wbf    = (__bf16*)d_ws;                       // 2 * W_LAYER
  __bf16* embwbf = wbf + (size_t)Ll * W_LAYER;          // 512*1536
  float*  biascat= (float*)(embwbf + (size_t)Dd * CATD);// 2 * 3072
  __bf16* kbf    = (__bf16*)(biascat + 2 * 2 * D3);
  __bf16* fbf0   = kbf + (size_t)Bb * Nn * Dd;
  __bf16* catbf  = fbf0 + (size_t)Bb * Nn * Dd;
  float*  Hf0    = (float*)(catbf + (size_t)Bb * Nn * CATD);
  float*  Hf1    = Hf0 + (size_t)Bb * Nn * Dd;
  __bf16* Hbf0   = (__bf16*)(Hf1 + (size_t)Bb * Nn * Dd);
  __bf16* Hbf1   = Hbf0 + (size_t)Bb * Nn * Dd;
  float*  gi_c   = (float*)(Hbf1 + (size_t)Bb * Nn * Dd);
  float*  gh_p   = gi_c + (size_t)Bb * Nn * D3;
  float*  gi_s   = gh_p + (size_t)Bb * Nn * D3;
  float*  gh_s   = gi_s + (size_t)Bb * Nn * D3;
  float*  hdot   = gh_s + (size_t)Bb * Nn * D3;
  __bf16* ucat   = (__bf16*)(hdot + (size_t)Bb * Nn);   // 16 x 1024 bf16
  float*  Mv     = (float*)(ucat + (size_t)Bb * 2 * Dd);
  __bf16* Mbf    = (__bf16*)(Mv + (size_t)Bb * Dd);
  float*  gbuf   = (float*)(Mbf + (size_t)Bb * Dd);     // 16 x 3072 f32

  dim3 cg(512), cb(256);
  // ---- one-time conversions / gathers ----
  for (int l = 0; l < Ll; ++l) {
    __bf16* wb = wbf + (size_t)l * W_LAYER;
    cvt_f32_bf16<<<cg, cb, 0, stream>>>(lin(l, L_C_WHH), wb + W_WHH_C, MW);
    cvt_f32_bf16<<<cg, cb, 0, stream>>>(lin(l, L_P_WIH), wb + W_WIH_P, MW);
    cvt_f32_bf16<<<cg, cb, 0, stream>>>(lin(l, L_C_WIH), wb + W_WIH_C, MW);
    cvt_f32_bf16<<<cg, cb, 0, stream>>>(lin(l, L_P_WHH), wb + W_WHH_P, MW);
    cvt_f32_bf16<<<cg, cb, 0, stream>>>(lin(l, L_S_WIH), wb + W_WIH_S, MW);
    cvt_f32_bf16<<<cg, cb, 0, stream>>>(lin(l, L_S_WHH), wb + W_WHH_S, MW);
    // wrcat: row n = [wr0[n, 0:512] | wr1[n, 0:512]]  (K-concat, ld = 1024)
    cvt_f32_bf16_strided<<<cg, cb, 0, stream>>>(lin(l, L_WR0), wb + W_WRCAT,       Dd, Dd, 2 * Dd);
    cvt_f32_bf16_strided<<<cg, cb, 0, stream>>>(lin(l, L_WR1), wb + W_WRCAT + Dd,  Dd, Dd, 2 * Dd);
    copy_f32<<<dim3(8), cb, 0, stream>>>(lin(l, L_C_BHH), biascat + (size_t)l * 2 * D3,      D3);
    copy_f32<<<dim3(8), cb, 0, stream>>>(lin(l, L_P_BIH), biascat + (size_t)l * 2 * D3 + D3, D3);
  }
  cvt_f32_bf16<<<cg, cb, 0, stream>>>(emb_w, embwbf, (size_t)Dd * CATD);
  gather_know_kernel<<<cg, cb, 0, stream>>>(knowledge, knowadj, kbf);
  pack_features_kernel<<<cg, cb, 0, stream>>>(features, fbf0, catbf);

  // ---- layers (sequential scan) ----
  for (int l = 0; l < Ll; ++l) {
    const __bf16* wb   = wbf + (size_t)l * W_LAYER;
    const float*  attw = lin(l, L_ATT_W);
    const float*  attb = lin(l, L_ATT_B);
    const float*  Fcur = (l == 0) ? features : Hf0;
    const __bf16* Fbf  = (l == 0) ? fbf0 : Hbf0;
    float*  Hc  = (l == 0) ? Hf0  : Hf1;
    __bf16* Hbc = (l == 0) ? Hbf0 : Hbf1;
    const float* bcat  = biascat + (size_t)l * 2 * D3;

    // state-independent GRU halves: four 1024x1536x512 WMMA GEMMs (NT=4)
    launch_gemm<4>(Fbf, wb + W_WIH_C, lin(l, L_C_BIH), gi_c, nullptr, Bb * Nn, D3, Dd, stream);
    launch_gemm<4>(Fbf, wb + W_WHH_P, lin(l, L_P_BHH), gh_p, nullptr, Bb * Nn, D3, Dd, stream);
    launch_gemm<4>(kbf, wb + W_WIH_S, lin(l, L_S_BIH), gi_s, nullptr, Bb * Nn, D3, Dd, stream);
    launch_gemm<4>(Fbf, wb + W_WHH_S, lin(l, L_S_BHH), gh_s, nullptr, Bb * Nn, D3, Dd, stream);

    init_step_kernel<<<dim3(Bb), cb, 0, stream>>>(gi_c, gh_p, gi_s, gh_s,
        lin(l, L_C_BHH), lin(l, L_P_BIH), Fcur, attw, Hc, Hbc, catbf, hdot, (l + 1) * Dd);

    for (int i = 1; i < Nn; ++i) {
      step_attn_kernel<<<dim3(Bb), cb, 0, stream>>>(Fcur, Hc, hdot, adj, smask, omask,
                                                    attw, attb, ucat, i);
      // Mvec = [u0|u1](16x1024) @ [wr0|wr1](512x1024)^T ; bf16 copy fused
      launch_gemm<2>(ucat, wb + W_WRCAT, nullptr, Mv, Mbf, Bb, Dd, 2 * Dd, stream);
      // gates: Mvec @ [w_hh_c ; w_ih_p]^T + [b_hh_c ; b_ih_p]  (16 x 3072, K=512)
      launch_gemm<2>(Mbf, wb + W_WHH_C, bcat, gbuf, nullptr, Bb, 2 * D3, Dd, stream);
      step_combine_kernel<<<dim3(Bb), cb, 0, stream>>>(gbuf, gi_c, gh_p, gi_s, gh_s,
          Mv, Fcur, attw, Hc, Hbc, catbf, hdot, i, (l + 1) * Dd);
    }
  }

  // ---- final embedding: cat(1024x1536) @ emb_w(512x1536)^T + emb_b ----
  launch_gemm<4>(catbf, embwbf, emb_b, (float*)d_out, nullptr, Bb * Nn, Dd, CATD, stream);
}
